// T5DenseReluDense_19241453486817
// MI455X (gfx1250) — matrix operand order repaired and checked by reading, not compiled, not run on hardware
//
#include <hip/hip_runtime.h>
#include <hip/hip_bf16.h>

typedef __bf16 bf16_t;
typedef __attribute__((ext_vector_type(16))) __bf16 v16bf;
typedef __attribute__((ext_vector_type(8)))  __bf16 v8bf;
typedef __attribute__((ext_vector_type(4)))  __bf16 v4bf;
typedef __attribute__((ext_vector_type(8)))  float  v8f;

#define D_MODEL 1024
#define D_FF    4096
#define TM      16            // tokens per workgroup
#define NWAVES  8             // 256 threads, wave32

// LDS layout (bytes)
#define HS_BYTES   (TM * D_FF * 2)        // 128 KB: h tile, bf16
#define XS_BYTES   (TM * D_MODEL * 2)     //  32 KB: x tile, bf16
#define HIST_BYTES (NWAVES * 256 * 4)     //   8 KB: per-wave radix histograms
#define SMEM_BYTES (HS_BYTES + XS_BYTES + HIST_BYTES)   // 168 KB <= 320 KB WGP LDS

#define WB_BYTES ((size_t)D_FF * D_MODEL * 2)   // one bf16 weight matrix = 8 MB

__device__ __forceinline__ v16bf cat8(v8bf lo, v8bf hi) {
  return __builtin_shufflevector(lo, hi, 0,1,2,3,4,5,6,7,8,9,10,11,12,13,14,15);
}

__device__ __forceinline__ v16bf cvt16(float4 b0, float4 b1, float4 b2, float4 b3) {
  v16bf b;
  b[0]  = (bf16_t)b0.x; b[1]  = (bf16_t)b0.y; b[2]  = (bf16_t)b0.z; b[3]  = (bf16_t)b0.w;
  b[4]  = (bf16_t)b1.x; b[5]  = (bf16_t)b1.y; b[6]  = (bf16_t)b1.z; b[7]  = (bf16_t)b1.w;
  b[8]  = (bf16_t)b2.x; b[9]  = (bf16_t)b2.y; b[10] = (bf16_t)b2.z; b[11] = (bf16_t)b2.w;
  b[12] = (bf16_t)b3.x; b[13] = (bf16_t)b3.y; b[14] = (bf16_t)b3.z; b[15] = (bf16_t)b3.w;
  return b;
}

// load a 32x16 B fragment: 16 contiguous K-values of one weight row per lane
template<bool W16>
__device__ __forceinline__ v16bf loadB(const float* wf, const bf16_t* wb, size_t off) {
  if constexpr (W16) {
    v8bf lo = *(const v8bf*)(wb + off);
    v8bf hi = *(const v8bf*)(wb + off + 8);
    return cat8(lo, hi);
  } else {
    const float4* bp = (const float4*)(wf + off);
    return cvt16(bp[0], bp[1], bp[2], bp[3]);
  }
}

// ------------- pre-pass: convert one f32 weight matrix to bf16 -------------
__global__ __launch_bounds__(256) void cvt_weights(const float* __restrict__ w,
                                                   bf16_t* __restrict__ o, int n4) {
  int i = blockIdx.x * 256 + threadIdx.x;
  if (i < n4) {
    float4 v = ((const float4*)w)[i];
    v4bf b; b[0]=(bf16_t)v.x; b[1]=(bf16_t)v.y; b[2]=(bf16_t)v.z; b[3]=(bf16_t)v.w;
    ((v4bf*)o)[i] = b;
  }
}

// ------------- fused FFN + top-k kernel -------------
template<bool W16>
__global__ __launch_bounds__(256) void ffn_topk_fused(
    const float* __restrict__ x, const float* __restrict__ wi,
    const float* __restrict__ wo, const bf16_t* __restrict__ wib,
    const bf16_t* __restrict__ wob, const int* __restrict__ topk,
    float* __restrict__ out)
{
  extern __shared__ char smem[];
  bf16_t*   hsb  = (bf16_t*)smem;                             // [TM][D_FF] bf16
  unsigned short* hu = (unsigned short*)smem;                 // bit view of hsb
  bf16_t*   xs   = (bf16_t*)(smem + HS_BYTES);                // [TM][D_MODEL]
  unsigned* hist = (unsigned*)(smem + HS_BYTES + XS_BYTES);   // [NWAVES][256]

  const int tid  = threadIdx.x;
  const int wave = tid >> 5;
  const int lane = tid & 31;
  const int half = lane >> 4;     // which 16-lane half of the wave
  const int l16  = lane & 15;
  const int t0   = blockIdx.x * TM;

  // ---------------- stage x tile -> LDS as bf16 ----------------
  for (int i = tid; i < TM * D_MODEL; i += 256) {
    int r = i >> 10, d = i & (D_MODEL - 1);
    xs[r * D_MODEL + d] = (bf16_t)x[(size_t)(t0 + r) * D_MODEL + d];
  }
  __syncthreads();

  // ---------------- GEMM1: h = relu(x @ wi^T) ----------------
  // wave covers d_ff columns [wave*512, wave*512+512): 8 groups of 4 tiles;
  // one A fragment feeds 4 wmmas per k-step.
  for (int tg = 0; tg < 32; tg += 4) {
    const int f0 = wave * 512 + tg * 16;
    v8f acc0 = {}, acc1 = {}, acc2 = {}, acc3 = {};
    for (int k0 = 0; k0 < D_MODEL; k0 += 32) {
      const int abase = l16 * D_MODEL + k0 + half * 16;
      v16bf a = cat8(*(const v8bf*)&xs[abase], *(const v8bf*)&xs[abase + 8]);
      const size_t bo = (size_t)(f0 + l16) * D_MODEL + k0 + half * 16;
      v16bf b0 = loadB<W16>(wi, wib, bo);
      v16bf b1 = loadB<W16>(wi, wib, bo + 16 * D_MODEL);
      v16bf b2 = loadB<W16>(wi, wib, bo + 32 * D_MODEL);
      v16bf b3 = loadB<W16>(wi, wib, bo + 48 * D_MODEL);
      acc0 = __builtin_amdgcn_wmma_f32_16x16x32_bf16(false, a, false, b0, (short)0, acc0, false, false);
      acc1 = __builtin_amdgcn_wmma_f32_16x16x32_bf16(false, a, false, b1, (short)0, acc1, false, false);
      acc2 = __builtin_amdgcn_wmma_f32_16x16x32_bf16(false, a, false, b2, (short)0, acc2, false, false);
      acc3 = __builtin_amdgcn_wmma_f32_16x16x32_bf16(false, a, false, b3, (short)0, acc3, false, false);
    }
#pragma unroll
    for (int r = 0; r < 8; ++r) {
      const size_t ro = (size_t)(r + half * 8) * D_FF + f0 + l16;
      float v0 = acc0[r], v1 = acc1[r], v2 = acc2[r], v3 = acc3[r];
      hsb[ro +  0] = (bf16_t)(v0 > 0.f ? v0 : 0.f);   // fused ReLU
      hsb[ro + 16] = (bf16_t)(v1 > 0.f ? v1 : 0.f);
      hsb[ro + 32] = (bf16_t)(v2 > 0.f ? v2 : 0.f);
      hsb[ro + 48] = (bf16_t)(v3 > 0.f ? v3 : 0.f);
    }
  }
  __syncthreads();

  // ---------------- top-k threshold per row (bf16 radix select) ----------------
  // post-ReLU bf16 values are >= 0: bit patterns order like unsigned ints.
  const int kkeep = topk[0];
  if (kkeep < D_FF) {
    unsigned* h8 = hist + wave * 256;              // private per wave
    for (int rr = 0; rr < 2; ++rr) {               // 8 waves x 2 rows = 16 rows
      const int row = wave * 2 + rr;
      unsigned short* hrow = hu + (size_t)row * D_FF;
      // level 1: histogram of bits [14:7] (exponent byte)
      for (int i = lane; i < 256; i += 32) h8[i] = 0;
      for (int i = lane; i < D_FF; i += 32)
        atomicAdd(&h8[hrow[i] >> 7], 1u);
      int b1 = 0; int c1 = 0;
      if (lane == 0) {
        unsigned cum = 0; int b = 255;
        for (; b > 0; --b) { unsigned h = h8[b]; if (cum + h >= (unsigned)kkeep) break; cum += h; }
        b1 = b; c1 = (int)cum;
      }
      b1 = __shfl(b1, 0); c1 = __shfl(c1, 0);
      const unsigned krem = (unsigned)(kkeep - c1);
      // level 2: histogram of bits [6:0] within exponent bucket b1
      for (int i = lane; i < 256; i += 32) h8[i] = 0;
      for (int i = lane; i < D_FF; i += 32) {
        unsigned u = hrow[i];
        if ((int)(u >> 7) == b1) atomicAdd(&h8[u & 127], 1u);
      }
      int b2 = 0;
      if (lane == 0) {
        unsigned cum = 0; int b = 127;
        for (; b > 0; --b) { unsigned h = h8[b]; if (cum + h >= krem) break; cum += h; }
        b2 = b;
      }
      b2 = __shfl(b2, 0);
      const unsigned T = ((unsigned)b1 << 7) | (unsigned)b2;
      // zero everything below threshold
      for (int i = lane; i < D_FF; i += 32)
        if (hrow[i] < T) hrow[i] = 0;
    }
  }
  __syncthreads();

  // ---------------- GEMM2: out = h_sparse @ wo^T ----------------
  // wave covers d_model columns [wave*128, wave*128+128): 2 groups of 4 tiles.
  for (int tg = 0; tg < 8; tg += 4) {
    const int m0 = wave * 128 + tg * 16;
    v8f acc0 = {}, acc1 = {}, acc2 = {}, acc3 = {};
    for (int k0 = 0; k0 < D_FF; k0 += 32) {
      const int abase = l16 * D_FF + k0 + half * 16;
      v16bf a = cat8(*(const v8bf*)&hsb[abase], *(const v8bf*)&hsb[abase + 8]);
      const size_t bo = (size_t)(m0 + l16) * D_FF + k0 + half * 16;
      v16bf b0 = loadB<W16>(wo, wob, bo);
      v16bf b1 = loadB<W16>(wo, wob, bo + 16 * D_FF);
      v16bf b2 = loadB<W16>(wo, wob, bo + 32 * D_FF);
      v16bf b3 = loadB<W16>(wo, wob, bo + 48 * D_FF);
      acc0 = __builtin_amdgcn_wmma_f32_16x16x32_bf16(false, a, false, b0, (short)0, acc0, false, false);
      acc1 = __builtin_amdgcn_wmma_f32_16x16x32_bf16(false, a, false, b1, (short)0, acc1, false, false);
      acc2 = __builtin_amdgcn_wmma_f32_16x16x32_bf16(false, a, false, b2, (short)0, acc2, false, false);
      acc3 = __builtin_amdgcn_wmma_f32_16x16x32_bf16(false, a, false, b3, (short)0, acc3, false, false);
    }
#pragma unroll
    for (int r = 0; r < 8; ++r) {
      const size_t ro = (size_t)(t0 + r + half * 8) * D_MODEL + m0 + l16;
      out[ro +  0] = acc0[r];
      out[ro + 16] = acc1[r];
      out[ro + 32] = acc2[r];
      out[ro + 48] = acc3[r];
    }
  }
}

extern "C" void kernel_launch(void* const* d_in, const int* in_sizes, int n_in,
                              void* d_out, int out_size, void* d_ws, size_t ws_size,
                              hipStream_t stream) {
  (void)n_in; (void)out_size;
  const float* x    = (const float*)d_in[0];
  const float* wi   = (const float*)d_in[1];
  const float* wo   = (const float*)d_in[2];
  const int*   topk = (const int*)d_in[3];
  float* out = (float*)d_out;

  const int ntok = in_sizes[0] / D_MODEL;   // 8192
  const int grid = ntok / TM;               // 512 workgroups
  const int n4   = D_FF * D_MODEL / 4;      // float4 elements per weight matrix

  const bool w16 = ws_size >= 2 * WB_BYTES; // need 16 MB of scratch for bf16 weights

  if (w16) {
    bf16_t* wib = (bf16_t*)d_ws;
    bf16_t* wob = wib + (size_t)D_FF * D_MODEL;
    cvt_weights<<<(n4 + 255) / 256, 256, 0, stream>>>(wi, wib, n4);
    cvt_weights<<<(n4 + 255) / 256, 256, 0, stream>>>(wo, wob, n4);
    (void)hipFuncSetAttribute((const void*)ffn_topk_fused<true>,
                              hipFuncAttributeMaxDynamicSharedMemorySize, SMEM_BYTES);
    ffn_topk_fused<true><<<grid, 256, SMEM_BYTES, stream>>>(x, wi, wo, wib, wob, topk, out);
  } else {
    (void)hipFuncSetAttribute((const void*)ffn_topk_fused<false>,
                              hipFuncAttributeMaxDynamicSharedMemorySize, SMEM_BYTES);
    ffn_topk_fused<false><<<grid, 256, SMEM_BYTES, stream>>>(x, wi, wo, (const bf16_t*)nullptr,
                                                             (const bf16_t*)nullptr, topk, out);
  }
}
